// MLP_EI_56779467653693
// MI455X (gfx1250) — matrix operand order repaired and checked by reading, not compiled, not run on hardware
//
#include <hip/hip_runtime.h>
#include <cstdint>

// CDNA5 (gfx1250) wave32 WMMA types
typedef __attribute__((ext_vector_type(16))) _Float16 v16h;
typedef __attribute__((ext_vector_type(8)))  float    v8f;
typedef __attribute__((ext_vector_type(4)))  int      v4i;

// gfx1250 async global->LDS (ASYNCcnt-tracked): (int4* AS1 gsrc, int4* AS3 ldst, imm, imm)
#if defined(__AMDGCN__) && __has_builtin(__builtin_amdgcn_global_load_async_to_lds_b128) && \
    __has_builtin(__builtin_amdgcn_s_wait_asynccnt)
#define USE_ASYNC_LDS 1
typedef __attribute__((address_space(1))) v4i as1_v4i;
typedef __attribute__((address_space(3))) v4i as3_v4i;
#define ASYNC_CP_B128(gsrc, ldst)                                                  \
    __builtin_amdgcn_global_load_async_to_lds_b128(                                \
        (as1_v4i*)(const void*)(gsrc), (as3_v4i*)(void*)(ldst), 0, 0)
#else
#define USE_ASYNC_LDS 0
#endif

#define BM 256          // block tile M (4 wave-rows x 64)
#define BN 128          // block tile N (2 wave-cols x 64)
#define TK 32           // K step per v_wmma_f32_16x16x32_f16
#define STR 40          // LDS row stride in halves: 80B rows (16B-aligned for async b128,
                        // 20-dword lane stride -> 16 lanes hit 16 distinct banks)
#define NE 3277         // round(4096 * 0.8)

union Frag { v16h v; uint32_t u[8]; };
union AccT { v8f  v; float    f[8]; };
union H8   { _Float16 h[8]; uint4 u4; };

// Out = epilogue( Ain(f16, MxK) @ Wt^T )  with Wt = +-1 f16, stored N-major (N x K)
// MODE 0: f16 Out = eif(col) * 2 * relu(scale * acc)   (eif: +1 col<NE, -4 else)
// MODE 1: f32 Out = scale * acc
template <int MODE>
__global__ __launch_bounds__(256)
void gemm_pm1_kernel(const _Float16* __restrict__ Ain,
                     const _Float16* __restrict__ Wt,
                     const float*    __restrict__ scale_ptr,
                     void*           __restrict__ OutV,
                     int M, int K, int N)
{
    __shared__ _Float16 sA[2][BM * STR];   // double-buffered A tile (40 KiB)
    __shared__ _Float16 sB[2][BN * STR];   // double-buffered B tile (20 KiB, transposed)

    const int tid  = threadIdx.x;
    const int lane = tid & 31;
    const int wave = tid >> 5;
    const int lh   = lane >> 4;
    const int ln   = lane & 15;

    const int blockN = blockIdx.x * BN;
    const int blockM = blockIdx.y * BM;

    // 4x2 wave grid -> 64x64 output tile per wave (4x4 accumulators, 16 WMMA : 16 ds)
    const int m0w = (wave >> 1) * 64;
    const int n0w = (wave & 1) * 64;

    const float scale = *scale_ptr;

    AccT acc[4][4];
#pragma unroll
    for (int mt = 0; mt < 4; ++mt)
#pragma unroll
        for (int nt = 0; nt < 4; ++nt)
#pragma unroll
            for (int r = 0; r < 8; ++r) acc[mt][nt].f[r] = 0.0f;

    // staging: A -> one 32-half row per thread (4 b128); B -> 16-half half-row (2 b128)
    const int aoff = tid * STR;
    const _Float16* agp = Ain + (size_t)(blockM + tid) * K;
    const int brow = tid >> 1;
    const int bseg = (tid & 1) * 16;
    const int boff = brow * STR + bseg;
    const _Float16* bgp = Wt + (size_t)(blockN + brow) * K + bseg;

    const int T = K / TK;

    // prologue: tile 0 in flight
#if USE_ASYNC_LDS
    ASYNC_CP_B128(agp,      &sA[0][aoff]);
    ASYNC_CP_B128(agp + 8,  &sA[0][aoff + 8]);
    ASYNC_CP_B128(agp + 16, &sA[0][aoff + 16]);
    ASYNC_CP_B128(agp + 24, &sA[0][aoff + 24]);
    ASYNC_CP_B128(bgp,      &sB[0][boff]);
    ASYNC_CP_B128(bgp + 8,  &sB[0][boff + 8]);
#else
    uint4 ar0 = *(const uint4*)agp,        ar1 = *(const uint4*)(agp + 8);
    uint4 ar2 = *(const uint4*)(agp + 16), ar3 = *(const uint4*)(agp + 24);
    uint4 br0 = *(const uint4*)bgp,        br1 = *(const uint4*)(bgp + 8);
#endif

    for (int t = 0; t < T; ++t) {
        const int cur = t & 1;
        const int kk  = t * TK;

#if USE_ASYNC_LDS
        __builtin_amdgcn_s_wait_asynccnt(0);   // this wave's tile-t copies landed in LDS
#else
        *(uint4*)&sA[cur][aoff]      = ar0;  *(uint4*)&sA[cur][aoff + 8]  = ar1;
        *(uint4*)&sA[cur][aoff + 16] = ar2;  *(uint4*)&sA[cur][aoff + 24] = ar3;
        *(uint4*)&sB[cur][boff]      = br0;  *(uint4*)&sB[cur][boff + 8]  = br1;
#endif
        __syncthreads();   // tile t staged by all waves; tile t-1 reads finished

        if (t + 1 < T) {   // overlap tile t+1 copy with tile t WMMA
#if USE_ASYNC_LDS
            ASYNC_CP_B128(agp + kk + TK,      &sA[1 - cur][aoff]);
            ASYNC_CP_B128(agp + kk + TK + 8,  &sA[1 - cur][aoff + 8]);
            ASYNC_CP_B128(agp + kk + TK + 16, &sA[1 - cur][aoff + 16]);
            ASYNC_CP_B128(agp + kk + TK + 24, &sA[1 - cur][aoff + 24]);
            ASYNC_CP_B128(bgp + kk + TK,      &sB[1 - cur][boff]);
            ASYNC_CP_B128(bgp + kk + TK + 8,  &sB[1 - cur][boff + 8]);
#else
            ar0 = *(const uint4*)(agp + kk + TK);
            ar1 = *(const uint4*)(agp + kk + TK + 8);
            ar2 = *(const uint4*)(agp + kk + TK + 16);
            ar3 = *(const uint4*)(agp + kk + TK + 24);
            br0 = *(const uint4*)(bgp + kk + TK);
            br1 = *(const uint4*)(bgp + kk + TK + 8);
#endif
            if (t + 2 < T) {   // global_prefetch_b8 two tiles ahead
                __builtin_prefetch(agp + kk + 2 * TK, 0, 1);
                __builtin_prefetch(bgp + kk + 2 * TK, 0, 1);
            }
        }

        // ---- fragments per CDNA5 VGPR layouts (ds_load_b128-able) ----
        // B 32x16: lane = n + 16*kh; v0..3 K = kh*8+{0..7}, v4..7 K = 16+kh*8+{0..7} of column n.
        // Load all 4 B frags once; stream A frags one at a time (low live-reg count).
        Frag b[4];
#pragma unroll
        for (int nt = 0; nt < 4; ++nt) {
            const int base = (n0w + nt * 16 + ln) * STR + lh * 8;
#pragma unroll
            for (int i = 0; i < 8; ++i)
                b[nt].u[i] = *(const uint32_t*)&sB[cur][base + 2 * i + (i >= 4 ? 8 : 0)];
        }
        // A 16x32: lane = m + 16*lh; v0..3 K = lh*8+{0..7}, v4..7 K = 16+lh*8+{0..7}
#pragma unroll
        for (int mt = 0; mt < 4; ++mt) {
            Frag a;
            const int base = (m0w + mt * 16 + ln) * STR;
#pragma unroll
            for (int i = 0; i < 4; ++i)
                a.u[i]     = *(const uint32_t*)&sA[cur][base + lh * 8 + 2 * i];
#pragma unroll
            for (int i = 0; i < 4; ++i)
                a.u[4 + i] = *(const uint32_t*)&sA[cur][base + 16 + lh * 8 + 2 * i];
#pragma unroll
            for (int nt = 0; nt < 4; ++nt)
                acc[mt][nt].v = __builtin_amdgcn_wmma_f32_16x16x32_f16(
                    false, a.v, false, b[nt].v,
                    (short)0, acc[mt][nt].v, false, false);
        }
    }

    // ---- epilogue: C/D layout -> VGPR r holds M = r + 8*lh, N = ln ----
#pragma unroll
    for (int mt = 0; mt < 4; ++mt) {
#pragma unroll
        for (int nt = 0; nt < 4; ++nt) {
            const int    gn  = blockN + n0w + nt * 16 + ln;
            const size_t row = (size_t)(blockM + m0w + mt * 16 + lh * 8) * N + gn;
            if (MODE == 0) {
                const float colf = (gn < NE) ? 2.0f : -8.0f;  // relu*2 and EI fold
                _Float16* Out = (_Float16*)OutV;
#pragma unroll
                for (int r = 0; r < 8; ++r) {
                    const float v = fmaxf(scale * acc[mt][nt].f[r], 0.0f) * colf;
                    Out[row + (size_t)r * N] = (_Float16)v;
                }
            } else {
                float* Out = (float*)OutV;
#pragma unroll
                for (int r = 0; r < 8; ++r)
                    Out[row + (size_t)r * N] = scale * acc[mt][nt].f[r];
            }
        }
    }
}

// bool W[k][n] -> +-1 f16 Wt[n][k]  (64x64 LDS tile transpose per block)
__global__ __launch_bounds__(256)
void wconv_kernel(const uint8_t* __restrict__ Wb, _Float16* __restrict__ Wt, int D)
{
    __shared__ uint8_t tile[64][80];   // 80B row stride: 16B-aligned vector stores
    const int nb = blockIdx.x * 64;    // n base
    const int kb = blockIdx.y * 64;    // k base
    const int r  = threadIdx.x >> 2;          // 0..63
    const int c  = (threadIdx.x & 3) * 16;    // 0,16,32,48

    *(uint4*)&tile[r][c] = *(const uint4*)(Wb + (size_t)(kb + r) * D + nb + c);
    __syncthreads();

    H8 o0, o1;
#pragma unroll
    for (int i = 0; i < 8; ++i) {
        o0.h[i] = (tile[c + i][r]     & 1) ? (_Float16)1.0f : (_Float16)(-1.0f);
        o1.h[i] = (tile[c + 8 + i][r] & 1) ? (_Float16)1.0f : (_Float16)(-1.0f);
    }
    _Float16* dst = Wt + (size_t)(nb + r) * D + kb + c;
    *(uint4*)dst       = o0.u4;
    *(uint4*)(dst + 8) = o1.u4;
}

// one-shot fp32 -> f16 conversion of x (8 elems/thread, b128 in / b128 out)
__global__ __launch_bounds__(256)
void f32_to_f16_kernel(const float* __restrict__ in, _Float16* __restrict__ out, size_t n)
{
    const size_t i = ((size_t)blockIdx.x * blockDim.x + threadIdx.x) * 8;
    if (i + 8 <= n) {
        const float4 a = *(const float4*)(in + i);
        const float4 b = *(const float4*)(in + i + 4);
        H8 o;
        o.h[0] = (_Float16)a.x; o.h[1] = (_Float16)a.y;
        o.h[2] = (_Float16)a.z; o.h[3] = (_Float16)a.w;
        o.h[4] = (_Float16)b.x; o.h[5] = (_Float16)b.y;
        o.h[6] = (_Float16)b.z; o.h[7] = (_Float16)b.w;
        *(uint4*)(out + i) = o.u4;
    }
}

extern "C" void kernel_launch(void* const* d_in, const int* in_sizes, int n_in,
                              void* d_out, int out_size, void* d_ws, size_t ws_size,
                              hipStream_t stream) {
    const float*   x  = (const float*)  d_in[0];
    const uint8_t* k0 = (const uint8_t*)d_in[1];   // numpy bool = 1 byte
    const float*   s0 = (const float*)  d_in[2];
    const uint8_t* k1 = (const uint8_t*)d_in[3];
    const float*   s1 = (const float*)  d_in[4];
    const uint8_t* k2 = (const uint8_t*)d_in[5];
    const float*   s2 = (const float*)  d_in[6];

    const int    D = 4096;
    const int    M = in_sizes[0] / D;              // 8192 tokens
    const size_t E = (size_t)M * D;

    _Float16* xf    = (_Float16*)d_ws;             // 64 MiB activations ping
    _Float16* h1    = xf + E;                      // 64 MiB activations pong
    _Float16* wslot = h1 + E;                      // 32 MiB transposed +-1 weights
    _Float16* h2    = xf;                          // reuse: x dead after L1

    dim3 block(256);
    dim3 ggrid(D / BN, M / BM);
    dim3 wgrid(D / 64, D / 64);

    f32_to_f16_kernel<<<dim3((unsigned)(E / (8 * 256))), block, 0, stream>>>(x, xf, E);

    // L1: h1 = f16( eif * 2*relu(s0 * x @ W0) )   (EI fold for layer-2 input)
    wconv_kernel<<<wgrid, block, 0, stream>>>(k0, wslot, D);
    gemm_pm1_kernel<0><<<ggrid, block, 0, stream>>>(xf, wslot, s0, (void*)h1, M, D, D);
    // L2: h2 = f16( eif * 2*relu(s1 * h1 @ W1) )  (EI fold for layer-3 input)
    wconv_kernel<<<wgrid, block, 0, stream>>>(k1, wslot, D);
    gemm_pm1_kernel<0><<<ggrid, block, 0, stream>>>(h1, wslot, s1, (void*)h2, M, D, D);
    // L3: out = f32( s2 * h2 @ W2 )
    wconv_kernel<<<wgrid, block, 0, stream>>>(k2, wslot, D);
    gemm_pm1_kernel<1><<<ggrid, block, 0, stream>>>(h2, wslot, s2, d_out, M, D, D);
}
